// GPTBlock_8211977470194
// MI455X (gfx1250) — compile-verified, matrix operand order checked by reading, not run on hardware
//
#include <hip/hip_runtime.h>
#include <hip/hip_bf16.h>

#define SEQ     4096
#define DMODEL  1024
#define NHEADS  2
#define HEADDIM 512
#define DFF     4096

typedef __attribute__((ext_vector_type(16))) __bf16 v16bf;
typedef __attribute__((ext_vector_type(8)))  __bf16 v8bf;
typedef __attribute__((ext_vector_type(8)))  float  v8f;

// 32-byte fragment load as two 16B halves (everything is 16B aligned by layout).
__device__ inline v16bf ld16(const __bf16* p) {
  union { v16bf v; v8bf h[2]; } u;
  u.h[0] = *(const v8bf*)(p);
  u.h[1] = *(const v8bf*)(p + 8);
  return u.v;
}

__device__ inline v8f wmma_bf16(v16bf a, v16bf b, v8f c) {
  // (neg_a, A, neg_b, B, c_mod, C, reuse_a, reuse_b)
  return __builtin_amdgcn_wmma_f32_16x16x32_bf16(false, a, false, b, (short)0, c,
                                                 false, false);
}

// gfx1250 async copy: 16B global -> LDS, tracked by ASYNCcnt (no VGPR data path).
// lds_off = low 32 bits of the generic pointer to __shared__ (LDS byte offset).
__device__ inline void async_cp16(unsigned lds_off, const __bf16* gsrc) {
  asm volatile("global_load_async_to_lds_b128 %0, %1, off"
               :: "v"(lds_off), "v"((unsigned long long)(uintptr_t)gsrc)
               : "memory");
}
__device__ inline void wait_async0() {
  asm volatile("s_wait_asynccnt 0" ::: "memory");
}

__device__ inline float gelu_tanh(float x) {
  const float c = 0.7978845608028654f;  // sqrt(2/pi)
  return 0.5f * x * (1.0f + tanhf(c * (x + 0.044715f * x * x * x)));
}

// ---------------------------------------------------------------------------
// f32 -> bf16 elementwise convert (weights, one-time per call)
// ---------------------------------------------------------------------------
__global__ __launch_bounds__(256) void cvt_f32_bf16(const float* __restrict__ in,
                                                    __bf16* __restrict__ out, int n) {
  for (int i = blockIdx.x * 256 + threadIdx.x; i < n; i += gridDim.x * 256)
    out[i] = (__bf16)in[i];
}

// ---------------------------------------------------------------------------
// LayerNorm over D_MODEL=1024, one row per block, bf16 output
// ---------------------------------------------------------------------------
__global__ __launch_bounds__(256) void layernorm_bf16(
    const float* __restrict__ x, const float* __restrict__ g,
    const float* __restrict__ b, __bf16* __restrict__ out) {
  const int row = blockIdx.x;
  const int tid = threadIdx.x;
  const float4 v = ((const float4*)(x + (size_t)row * DMODEL))[tid];
  __shared__ float rs[256], rs2[256];
  rs[tid]  = v.x + v.y + v.z + v.w;
  rs2[tid] = v.x * v.x + v.y * v.y + v.z * v.z + v.w * v.w;
  __syncthreads();
  for (int off = 128; off > 0; off >>= 1) {
    if (tid < off) { rs[tid] += rs[tid + off]; rs2[tid] += rs2[tid + off]; }
    __syncthreads();
  }
  const float mu  = rs[0] * (1.0f / DMODEL);
  const float var = rs2[0] * (1.0f / DMODEL) - mu * mu;
  const float inv = rsqrtf(var + 1e-5f);
  const float4 gg = ((const float4*)g)[tid];
  const float4 bb = ((const float4*)b)[tid];
  __bf16* op = out + (size_t)row * DMODEL + tid * 4;
  op[0] = (__bf16)((v.x - mu) * inv * gg.x + bb.x);
  op[1] = (__bf16)((v.y - mu) * inv * gg.y + bb.y);
  op[2] = (__bf16)((v.z - mu) * inv * gg.z + bb.z);
  op[3] = (__bf16)((v.w - mu) * inv * gg.w + bb.w);
}

// ---------------------------------------------------------------------------
// NT GEMM:  Y(MxN) = A(MxK, row-major) * B(NxK, row-major)^T  via bf16 WMMA.
// Block: 256 threads = 8 waves; tile 128(M) x 128(N), BK=64.
// Wave w owns rows [w*16, w*16+16) x all 128 N columns: 8 f32 accumulators,
// 16 WMMAs per k-step (A-fragment amortized over 8 N-tiles).
// Double-buffered LDS tiles filled with global_load_async_to_lds_b128 so the
// next tile's DMA overlaps the current tile's WMMAs (one barrier per k-step).
// Padded stride 72 (x2B = 144B): 16-row fragment reads are bank-conflict-free.
// Epilogue flags: bias add, tanh-GELU, fp32 residual, fp32 out, bf16 out,
// bf16 transposed out (used to produce V^T for the attention PV GEMM).
// ---------------------------------------------------------------------------
#define TSTRIDE 72

__device__ inline void stage_tile_async(const __bf16* __restrict__ A,
                                        const __bf16* __restrict__ B,
                                        __bf16* as, __bf16* bs,
                                        int m0, int n0, int K, int k0, int tid) {
  const int lr = tid >> 1;            // 0..127 : tile row
  const int lh = (tid & 1) * 32;      // which 32-element half of the 64-wide row
  {
    const __bf16* src = A + (size_t)(m0 + lr) * K + k0 + lh;
    const unsigned dst = (unsigned)(uintptr_t)&as[lr * TSTRIDE + lh];
    async_cp16(dst,      src);
    async_cp16(dst + 16, src + 8);
    async_cp16(dst + 32, src + 16);
    async_cp16(dst + 48, src + 24);
  }
  {
    const __bf16* src = B + (size_t)(n0 + lr) * K + k0 + lh;
    const unsigned dst = (unsigned)(uintptr_t)&bs[lr * TSTRIDE + lh];
    async_cp16(dst,      src);
    async_cp16(dst + 16, src + 8);
    async_cp16(dst + 32, src + 16);
    async_cp16(dst + 48, src + 24);
  }
}

template <bool BIAS, bool DO_GELU, bool RES, bool OUTF, bool OUTB, bool OUTBT>
__global__ __launch_bounds__(256) void gemm_bf16_nt(
    const __bf16* __restrict__ A, const __bf16* __restrict__ B,
    const float* __restrict__ bias, const float* __restrict__ res,
    float* __restrict__ outf, __bf16* __restrict__ outb,
    int M, int N, int K) {
  __shared__ __align__(16) __bf16 As[2][128 * TSTRIDE];  // 2 x 18 KB
  __shared__ __align__(16) __bf16 Bs[2][128 * TSTRIDE];  // 2 x 18 KB

  const int tid   = threadIdx.x;
  const int wave  = tid >> 5;
  const int lane  = tid & 31;
  const int cn    = lane & 15;   // N index within 16x16 tile / A row
  const int khalf = lane >> 4;   // which K-half of fragment
  const int mhi   = khalf * 8;   // accumulator row offset

  const int m0 = blockIdx.y * 128;
  const int n0 = blockIdx.x * 128;

  v8f acc[8];
  for (int i = 0; i < 8; ++i)
    for (int r = 0; r < 8; ++r) acc[i][r] = 0.0f;

  // prologue: stage first tile
  stage_tile_async(A, B, As[0], Bs[0], m0, n0, K, 0, tid);
  wait_async0();
  __syncthreads();

  for (int k0 = 0; k0 < K; k0 += 64) {
    const int cur = (k0 >> 6) & 1;
    if (k0 + 64 < K)  // prefetch next tile into the other buffer (overlaps WMMAs)
      stage_tile_async(A, B, As[cur ^ 1], Bs[cur ^ 1], m0, n0, K, k0 + 64, tid);

#pragma unroll
    for (int kf = 0; kf < 2; ++kf) {
      v16bf af = ld16(&As[cur][(wave * 16 + cn) * TSTRIDE + kf * 32 + khalf * 16]);
#pragma unroll
      for (int nt = 0; nt < 8; ++nt) {
        v16bf bf = ld16(&Bs[cur][(nt * 16 + cn) * TSTRIDE + kf * 32 + khalf * 16]);
        acc[nt] = wmma_bf16(af, bf, acc[nt]);
      }
    }

    wait_async0();   // our own prefetch done
    __syncthreads(); // everyone's prefetch visible; cur buffer free for reuse
  }

#pragma unroll
  for (int nt = 0; nt < 8; ++nt) {
    const int gn = n0 + nt * 16 + cn;
    float bv = 0.0f;
    if (BIAS) bv = bias[gn];
#pragma unroll
    for (int r = 0; r < 8; ++r) {
      const int gm = m0 + wave * 16 + mhi + r;
      float v = acc[nt][r] + bv;
      if (DO_GELU) v = gelu_tanh(v);
      if (RES) v += res[(size_t)gm * N + gn];
      if (OUTF)  outf[(size_t)gm * N + gn] = v;
      if (OUTB)  outb[(size_t)gm * N + gn] = (__bf16)v;
      if (OUTBT) outb[(size_t)gn * M + gm] = (__bf16)v;  // transposed (V^T)
    }
  }
}

// ---------------------------------------------------------------------------
// Flash attention, one block per (head, 16-query tile); 8 waves.
// Per 128-key block: wave w computes a 16x16 Q*K^T tile (K=512, 16 WMMAs),
// LDS online softmax, then each wave accumulates 64 head-dim columns of
// P(16x128) * V (K=128, 16 WMMAs) using the pre-transposed V^T.
// ---------------------------------------------------------------------------
__global__ __launch_bounds__(256) void attn_flash(
    const __bf16* __restrict__ q, const __bf16* __restrict__ kk,
    const __bf16* __restrict__ vT, __bf16* __restrict__ o) {
  const int tid   = threadIdx.x;
  const int wave  = tid >> 5;
  const int lane  = tid & 31;
  const int cn    = lane & 15;
  const int khalf = lane >> 4;
  const int mhi   = khalf * 8;

  const int head  = blockIdx.x >> 8;  // SEQ/16 = 256 q-tiles per head
  const int qrow0 = (blockIdx.x & 255) * 16;
  const size_t hoff = (size_t)head * HEADDIM;

  __shared__ __align__(16) float  S_s[16 * 132];
  __shared__ __align__(16) __bf16 P_s[16 * 136];
  __shared__ float red[16][16];
  __shared__ float rowmax[16], rowsum[16], alpha_s[16];

  if (tid < 16) { rowmax[tid] = -3.0e38f; rowsum[tid] = 0.0f; }
  v8f oacc[4];
#pragma unroll
  for (int i = 0; i < 4; ++i)
#pragma unroll
    for (int r = 0; r < 8; ++r) oacc[i][r] = 0.0f;
  __syncthreads();

  const float inv_scale = 0.04419417382415922f;  // 1/sqrt(512)
  const int row16 = tid >> 4;
  const int col16 = tid & 15;

  for (int kb = 0; kb < SEQ; kb += 128) {
    // ---- scores: Q(16xK) * K^T ----
    const int key0 = kb + wave * 16;
    v8f sacc;
#pragma unroll
    for (int r = 0; r < 8; ++r) sacc[r] = 0.0f;
    const __bf16* qbase = q  + (size_t)(qrow0 + cn) * DMODEL + hoff + khalf * 16;
    const __bf16* kbase = kk + (size_t)(key0 + cn)  * DMODEL + hoff + khalf * 16;
#pragma unroll 4
    for (int ks = 0; ks < HEADDIM / 32; ++ks) {
      v16bf af = ld16(qbase + ks * 32);
      v16bf bf = ld16(kbase + ks * 32);
      sacc = wmma_bf16(af, bf, sacc);
    }
#pragma unroll
    for (int r = 0; r < 8; ++r)
      S_s[(mhi + r) * 132 + wave * 16 + cn] = sacc[r] * inv_scale;
    __syncthreads();

    // ---- online softmax: row max ----
    float m = -3.0e38f;
#pragma unroll
    for (int j = col16; j < 128; j += 16) m = fmaxf(m, S_s[row16 * 132 + j]);
    red[row16][col16] = m;
    __syncthreads();
    if (tid < 16) {
      float mm = red[tid][0];
#pragma unroll
      for (int j = 1; j < 16; ++j) mm = fmaxf(mm, red[tid][j]);
      const float mo = rowmax[tid];
      const float mn = fmaxf(mo, mm);
      alpha_s[tid] = __expf(mo - mn);
      rowmax[tid]  = mn;
    }
    __syncthreads();

    // ---- exp, P (bf16), row sum ----
    {
      const float mn = rowmax[row16];
      float s = 0.0f;
#pragma unroll
      for (int j = col16; j < 128; j += 16) {
        const float p = __expf(S_s[row16 * 132 + j] - mn);
        P_s[row16 * 136 + j] = (__bf16)p;
        s += p;
      }
      red[row16][col16] = s;
    }
    __syncthreads();
    if (tid < 16) {
      float s = 0.0f;
#pragma unroll
      for (int j = 0; j < 16; ++j) s += red[tid][j];
      rowsum[tid] = rowsum[tid] * alpha_s[tid] + s;
    }
    __syncthreads();

    // ---- rescale running O, then P*V ----
#pragma unroll
    for (int r = 0; r < 8; ++r) {
      const float a = alpha_s[mhi + r];
#pragma unroll
      for (int nt = 0; nt < 4; ++nt) oacc[nt][r] *= a;
    }
#pragma unroll
    for (int ks = 0; ks < 4; ++ks) {
      v16bf af = ld16(&P_s[cn * 136 + khalf * 16 + ks * 32]);
#pragma unroll
      for (int nt = 0; nt < 4; ++nt) {
        const int n = wave * 64 + nt * 16 + cn;
        v16bf bf = ld16(vT + (hoff + n) * (size_t)SEQ + kb + ks * 32 + khalf * 16);
        oacc[nt] = wmma_bf16(af, bf, oacc[nt]);
      }
    }
    __syncthreads();
  }

  // ---- normalize and write (bf16, row-major [S, D_MODEL]) ----
#pragma unroll
  for (int r = 0; r < 8; ++r) {
    const int mrow = mhi + r;
    const float inv = 1.0f / rowsum[mrow];
#pragma unroll
    for (int nt = 0; nt < 4; ++nt) {
      const int n = wave * 64 + nt * 16 + cn;
      o[(size_t)(qrow0 + mrow) * DMODEL + hoff + n] = (__bf16)(oacc[nt][r] * inv);
    }
  }
}

// ---------------------------------------------------------------------------
extern "C" void kernel_launch(void* const* d_in, const int* in_sizes, int n_in,
                              void* d_out, int out_size, void* d_ws, size_t ws_size,
                              hipStream_t stream) {
  (void)in_sizes; (void)n_in; (void)out_size; (void)ws_size;

  const float* x     = (const float*)d_in[0];
  const float* ln1_g = (const float*)d_in[1];
  const float* ln1_b = (const float*)d_in[2];
  const float* wq    = (const float*)d_in[3];
  const float* bq    = (const float*)d_in[4];
  const float* wk    = (const float*)d_in[5];
  const float* bk    = (const float*)d_in[6];
  const float* wv    = (const float*)d_in[7];
  const float* bv    = (const float*)d_in[8];
  const float* wo    = (const float*)d_in[9];
  const float* bo    = (const float*)d_in[10];
  const float* ln2_g = (const float*)d_in[11];
  const float* ln2_b = (const float*)d_in[12];
  const float* w1    = (const float*)d_in[13];
  const float* b1    = (const float*)d_in[14];
  const float* w2    = (const float*)d_in[15];
  const float* b2    = (const float*)d_in[16];
  float* out = (float*)d_out;

  char* ws = (char*)d_ws;
  size_t off = 0;
  auto carve = [&](size_t bytes) -> char* {
    char* p = ws + off;
    off += (bytes + 255) & ~(size_t)255;
    return p;
  };
  __bf16* wq_bf  = (__bf16*)carve((size_t)DMODEL * DMODEL * 2);
  __bf16* wk_bf  = (__bf16*)carve((size_t)DMODEL * DMODEL * 2);
  __bf16* wv_bf  = (__bf16*)carve((size_t)DMODEL * DMODEL * 2);
  __bf16* wo_bf  = (__bf16*)carve((size_t)DMODEL * DMODEL * 2);
  __bf16* w1_bf  = (__bf16*)carve((size_t)DFF * DMODEL * 2);
  __bf16* w2_bf  = (__bf16*)carve((size_t)DMODEL * DFF * 2);
  __bf16* h_bf   = (__bf16*)carve((size_t)SEQ * DMODEL * 2);  // reused for LN2
  __bf16* q_bf   = (__bf16*)carve((size_t)SEQ * DMODEL * 2);
  __bf16* k_bf   = (__bf16*)carve((size_t)SEQ * DMODEL * 2);
  __bf16* vT_bf  = (__bf16*)carve((size_t)SEQ * DMODEL * 2);  // [d_model][seq]
  __bf16* o_bf   = (__bf16*)carve((size_t)SEQ * DMODEL * 2);
  __bf16* ffa_bf = (__bf16*)carve((size_t)SEQ * DFF * 2);

  const dim3 blk(256);

  // one-time bf16 weight copies (L2-resident afterwards)
  cvt_f32_bf16<<<2048, blk, 0, stream>>>(wq, wq_bf, DMODEL * DMODEL);
  cvt_f32_bf16<<<2048, blk, 0, stream>>>(wk, wk_bf, DMODEL * DMODEL);
  cvt_f32_bf16<<<2048, blk, 0, stream>>>(wv, wv_bf, DMODEL * DMODEL);
  cvt_f32_bf16<<<2048, blk, 0, stream>>>(wo, wo_bf, DMODEL * DMODEL);
  cvt_f32_bf16<<<2048, blk, 0, stream>>>(w1, w1_bf, DFF * DMODEL);
  cvt_f32_bf16<<<2048, blk, 0, stream>>>(w2, w2_bf, DMODEL * DFF);

  // h = LN1(x)  (bf16)
  layernorm_bf16<<<SEQ, blk, 0, stream>>>(x, ln1_g, ln1_b, h_bf);

  const dim3 gD(DMODEL / 128, SEQ / 128);  // N=1024 GEMMs
  const dim3 gF(DFF / 128,    SEQ / 128);  // N=4096 GEMM

  // q, k (row-major bf16), v (transposed bf16)
  gemm_bf16_nt<true, false, false, false, true, false><<<gD, blk, 0, stream>>>(
      h_bf, wq_bf, bq, nullptr, nullptr, q_bf, SEQ, DMODEL, DMODEL);
  gemm_bf16_nt<true, false, false, false, true, false><<<gD, blk, 0, stream>>>(
      h_bf, wk_bf, bk, nullptr, nullptr, k_bf, SEQ, DMODEL, DMODEL);
  gemm_bf16_nt<true, false, false, false, false, true><<<gD, blk, 0, stream>>>(
      h_bf, wv_bf, bv, nullptr, nullptr, vT_bf, SEQ, DMODEL, DMODEL);

  // flash attention
  attn_flash<<<dim3(NHEADS * (SEQ / 16)), blk, 0, stream>>>(q_bf, k_bf, vT_bf, o_bf);

  // x1 = x + o @ Wo^T + bo   (fp32 -> d_out)
  gemm_bf16_nt<true, false, true, true, false, false><<<gD, blk, 0, stream>>>(
      o_bf, wo_bf, bo, x, out, nullptr, SEQ, DMODEL, DMODEL);

  // h2 = LN2(x1)  (bf16, reuse h_bf)
  layernorm_bf16<<<SEQ, blk, 0, stream>>>(out, ln2_g, ln2_b, h_bf);

  // ffa = gelu(h2 @ W1^T + b1)  (bf16)
  gemm_bf16_nt<true, true, false, false, true, false><<<gF, blk, 0, stream>>>(
      h_bf, w1_bf, b1, nullptr, nullptr, ffa_bf, SEQ, DFF, DMODEL);

  // out = x1 + ffa @ W2^T + b2  (fp32, in-place read/write of d_out per-element)
  gemm_bf16_nt<true, false, true, true, false, false><<<gD, blk, 0, stream>>>(
      ffa_bf, w2_bf, b2, out, out, nullptr, SEQ, DMODEL, DFF);
}
